// DCTPatchEmbedder_31258771980423
// MI455X (gfx1250) — compile-verified, matrix operand order checked by reading, not run on hardware
//
#include <hip/hip_runtime.h>

// ---------------------------------------------------------------------------
// DCT patch embedder collapsed into a single f16 WMMA GEMM.
//   out[m, n] = sum_k pix[m, k] * W_eff[k, n] + bias[n]
//   M = 16*64*64 = 65536 patches, K = 4*8*8 = 256 pixels, N = 1152
// W_eff folds (2D-DCT x zigzag-select x proj_w) and is built once per launch
// in f16 into d_ws (576 KB), transposed (N-major, K-contiguous).
// ---------------------------------------------------------------------------

typedef _Float16 h4   __attribute__((ext_vector_type(4)));
typedef _Float16 h8   __attribute__((ext_vector_type(8)));
typedef _Float16 v16h __attribute__((ext_vector_type(16)));
typedef float    v8f  __attribute__((ext_vector_type(8)));

#define INCH   4
#define EMBED  1152
#define HF     48
#define KTOT   256            // INCH * 64 pixels per patch
#define BM     64             // M rows per workgroup
#define BN     64             // N cols per workgroup
#define LDK    (KTOT + 8)     // padded LDS row (528 B -> +4 bank shift per row)

// Last 48 entries of the JPEG zigzag order (the kept coefficients).
__constant__ int ZZ48[48] = {
  12,19,26,33,40,48,41,34,27,20,13, 6, 7,14,21,28,
  35,42,49,56,57,50,43,36,29,22,15,23,30,37,44,51,
  58,59,52,45,38,31,39,46,53,60,61,54,47,55,62,63
};

__device__ __forceinline__ float dct8(int k, int i) {
  // Orthonormal DCT-II basis, N = 8: s_k * cos(pi*(2i+1)*k / 16)
  const float s = (k == 0) ? 0.35355339059327373f : 0.5f;
  return s * __cosf(0.19634954084936207f * (2.0f * (float)i + 1.0f) * (float)k);
}

// ---------------------------------------------------------------------------
// Prep: W_eff^T[n][kk] (f16), kk = c*64 + i*8 + j
//   W_eff[kk, n] = sum_t D[kz_t, i] * D[lz_t, j] * proj_w[c*48 + t, n]
// 1152 * 256 elements, 48-term reduction each — negligible cost.
// ---------------------------------------------------------------------------
__global__ __launch_bounds__(256)
void prep_weights(const float* __restrict__ proj_w, _Float16* __restrict__ wt) {
  int idx = blockIdx.x * blockDim.x + threadIdx.x;     // n*256 + kk
  if (idx >= EMBED * KTOT) return;
  const int n  = idx >> 8;
  const int kk = idx & 255;
  const int c  = kk >> 6;
  const int i  = (kk >> 3) & 7;
  const int j  = kk & 7;
  float acc = 0.0f;
#pragma unroll 8
  for (int t = 0; t < HF; ++t) {
    const int z  = ZZ48[t];
    const int kz = z >> 3;
    const int lz = z & 7;
    acc = fmaf(dct8(kz, i) * dct8(lz, j), proj_w[(c * HF + t) * EMBED + n], acc);
  }
  wt[n * KTOT + kk] = (_Float16)acc;
}

// ---------------------------------------------------------------------------
// GEMM: 128 threads = 4 waves. Block tile 64x64, wave tile 16x64.
// K = 256 staged fully in LDS (A gathered+converted from x, B raw f16 copy),
// then 8 K-steps of v_wmma_f32_16x16x32_f16 per N-tile.
// ---------------------------------------------------------------------------
__global__ __launch_bounds__(128)
void dct_embed_gemm(const float*    __restrict__ x,
                    const _Float16* __restrict__ wt,
                    const float*    __restrict__ bias,
                    float*          __restrict__ out) {
  __shared__ _Float16 As[BM][LDK];   // patch pixels, K-contiguous per M-row
  __shared__ _Float16 Bs[BN][LDK];   // W_eff^T,      K-contiguous per N-row

  const int tid  = threadIdx.x;
  const int lane = tid & 31;
  const int wave = tid >> 5;

  const int n0 = blockIdx.x * BN;    // 0..1151, 18 blocks
  const int m0 = blockIdx.y * BM;    // 0..65535, 1024 blocks

  // ---- stage B: 64 rows x 256 f16 = 2048 x 16B chunks, 16 per thread ----
#pragma unroll
  for (int it = 0; it < 16; ++it) {
    const int chunk = it * 128 + tid;          // 0..2047
    const int row   = chunk >> 5;              // n_local
    const int kc    = (chunk & 31) * 8;        // f16 offset, 16B aligned
    const uint4 v = *(const uint4*)(wt + (size_t)(n0 + row) * KTOT + kc);
    *(uint4*)(&Bs[row][kc]) = v;
  }

  // ---- stage A: gather patch pixels, f32 -> f16. 4096 float4 chunks ----
#pragma unroll
  for (int it = 0; it < 32; ++it) {
    const int chunk = it * 128 + tid;          // 0..4095
    const int row   = chunk >> 6;              // m_local
    const int kv    = (chunk & 63) * 4;        // kk base (j in {0,4})
    const int m  = m0 + row;
    const int b  = m >> 12;                    // batch
    const int p  = m & 4095;                   // patch in image
    const int ph = p >> 6;
    const int pw = p & 63;
    const int c  = kv >> 6;
    const int r  = (kv >> 3) & 7;              // i (row in patch)
    const int j  = kv & 7;
    const size_t gidx = ((size_t)(b * INCH + c) * 512 + (size_t)(ph * 8 + r)) * 512
                      + (size_t)(pw * 8 + j);
    const float4 v = *(const float4*)(x + gidx);
    h4 hv = { (_Float16)v.x, (_Float16)v.y, (_Float16)v.z, (_Float16)v.w };
    *(h4*)(&As[row][kv]) = hv;
  }

  __syncthreads();

  // ---- WMMA accumulation over K ----
  v8f acc[4];
#pragma unroll
  for (int t = 0; t < 4; ++t) acc[t] = (v8f){0.f,0.f,0.f,0.f,0.f,0.f,0.f,0.f};

  // ISA 16-bit operand layout (wave32):
  //   lanes 0-15 : row = lane,    K chunks [0..7] and [16..23]
  //   lanes 16-31: row = lane-16, K chunks [8..15] and [24..31]
  const int arow  = wave * 16 + (lane & 15);
  const int koff0 = (lane < 16) ? 0 : 8;
  const int koff1 = koff0 + 16;

#pragma unroll
  for (int ks = 0; ks < KTOT; ks += 32) {
    const h8 a_lo = *(const h8*)(&As[arow][ks + koff0]);
    const h8 a_hi = *(const h8*)(&As[arow][ks + koff1]);
    const v16h a = __builtin_shufflevector(a_lo, a_hi,
        0,1,2,3,4,5,6,7,8,9,10,11,12,13,14,15);
#pragma unroll
    for (int t = 0; t < 4; ++t) {
      const int brow = t * 16 + (lane & 15);
      const h8 b_lo = *(const h8*)(&Bs[brow][ks + koff0]);
      const h8 b_hi = *(const h8*)(&Bs[brow][ks + koff1]);
      const v16h bf = __builtin_shufflevector(b_lo, b_hi,
          0,1,2,3,4,5,6,7,8,9,10,11,12,13,14,15);
      // (neg_a, A, neg_b, B, c_mod, C, reuse_a, reuse_b)
      acc[t] = __builtin_amdgcn_wmma_f32_16x16x32_f16(
          false, a, false, bf, (short)0, acc[t], false, false);
    }
  }

  // ---- epilogue: bias + store (C/D layout: lane = N%16 + 16*(M>=8)) ----
  const int mloc = wave * 16 + 8 * (lane >> 4);   // + r gives M within block
  const int ncol = lane & 15;
#pragma unroll
  for (int t = 0; t < 4; ++t) {
    const int   n    = n0 + t * 16 + ncol;
    const float bval = bias[n];
#pragma unroll
    for (int r = 0; r < 8; ++r) {
      const size_t m = (size_t)(m0 + mloc + r);
      out[m * EMBED + n] = acc[t][r] + bval;
    }
  }
}

// ---------------------------------------------------------------------------
extern "C" void kernel_launch(void* const* d_in, const int* in_sizes, int n_in,
                              void* d_out, int out_size, void* d_ws, size_t ws_size,
                              hipStream_t stream) {
  const float* x  = (const float*)d_in[0];   // (16, 4, 512, 512) f32
  const float* pw = (const float*)d_in[1];   // (192, 1152) f32
  const float* pb = (const float*)d_in[2];   // (1152,) f32
  float*    out   = (float*)d_out;           // (16, 4096, 1152) f32
  _Float16* wt    = (_Float16*)d_ws;         // W_eff^T: 1152 x 256 f16 (576 KB)

  prep_weights<<<(EMBED * KTOT + 255) / 256, 256, 0, stream>>>(pw, wt);

  dim3 grid(EMBED / BN, (16 * 64 * 64) / BM);  // (18, 1024)
  dct_embed_gemm<<<grid, 128, 0, stream>>>(x, wt, pb, out);
}
